// graph_encoder_53764400611704
// MI455X (gfx1250) — compile-verified, matrix operand order checked by reading, not run on hardware
//
#include <hip/hip_runtime.h>

#define GN 20000      // nodes
#define GK 256        // x feature dim (= K)
#define GH 8          // heads
#define GD 128        // per-head dim
#define GHD 1024      // H*D
#define GCOLS 2048    // fused output cols (W | res_W)
#define GZ 64         // latent dim

#define BM 32         // block tile M
#define BN 256        // block tile N
#define LDSK 264      // padded LDS row stride in elements (256 + 8)

typedef __attribute__((ext_vector_type(16))) __bf16 v16bf;
typedef __attribute__((ext_vector_type(8)))  __bf16 v8bf;
typedef __attribute__((ext_vector_type(8)))  float  v8f;
typedef unsigned int v4u __attribute__((ext_vector_type(4)));
typedef int          v8i __attribute__((ext_vector_type(8)));
typedef int          v4i __attribute__((ext_vector_type(4)));

// ---------------- prep kernels ----------------

__global__ void convert_x_kernel(const float* __restrict__ x, __bf16* __restrict__ Xb, int n) {
    int i = blockIdx.x * blockDim.x + threadIdx.x;
    if (i < n) Xb[i] = (__bf16)x[i];
}

// BcatT[col][k] : pre-transposed fused weights so WMMA B-fragments are contiguous per lane
__global__ void convert_B_kernel(const float* __restrict__ W, const float* __restrict__ resW,
                                 __bf16* __restrict__ BcatT) {
    int i = blockIdx.x * blockDim.x + threadIdx.x;
    if (i >= GK * GCOLS) return;
    int col = i / GK, k = i % GK;
    float v = (col < GHD) ? W[k * GHD + col] : resW[k * GHD + (col - GHD)];
    BcatT[(size_t)col * GK + k] = (__bf16)v;
}

__global__ void init_kernel(float* __restrict__ m, float* __restrict__ s, float* __restrict__ stats) {
    int i = blockIdx.x * blockDim.x + threadIdx.x;
    if (i < GN * GH) { m[i] = -INFINITY; s[i] = 0.0f; }
    if (i < 2 * GD) stats[i] = 0.0f;
}

// ---------------- TDM tile load (2-D, bf16, padded LDS rows) ----------------
// D# per ISA 8.3/8.4: group0 = {count=1 | lds_addr | global_addr | type=2},
// group1 = {data_size=2B, pad_enable, pad_interval=128dw, pad_amount=4dw,
//           tensor_dim0/1, tile_dim0/1, tensor_dim0_stride}

#if __has_builtin(__builtin_amdgcn_tensor_load_to_lds)
#define HAVE_TDM 1
__device__ __forceinline__ void tdm_load_2d(unsigned lds_off, const void* gaddr,
                                            unsigned tile_k, unsigned tile_rows,
                                            unsigned tensor_d0, unsigned tensor_d1,
                                            unsigned stride_elems) {
    unsigned long long ga = (unsigned long long)(uintptr_t)gaddr;
    v4u g0;
    g0[0] = 1u;                                        // count=1, user descriptor
    g0[1] = lds_off;                                   // lds_addr (bytes)
    g0[2] = (unsigned)(ga & 0xffffffffu);              // global_addr[31:0]
    g0[3] = (unsigned)((ga >> 32) & 0x01ffffffu) | (2u << 30);  // global_addr[56:32] | type=2
    v8i g1;
    g1[0] = (int)((1u << 16) | (1u << 20) | (6u << 22) | (3u << 25));
        // data_size=2B | pad_enable | pad_interval=128 dwords | pad_amount=4 dwords
    g1[1] = (int)((tensor_d0 & 0xffffu) << 16);        // tensor_dim0[15:0]
    g1[2] = (int)(((tensor_d0 >> 16) & 0xffffu) | ((tensor_d1 & 0xffffu) << 16));
    g1[3] = (int)(((tensor_d1 >> 16) & 0xffffu) | ((tile_k & 0xffffu) << 16));   // tile_dim0
    g1[4] = (int)(tile_rows & 0xffffu);                // tile_dim1 (tile_dim2 = 0)
    g1[5] = (int)stride_elems;                         // tensor_dim0_stride[31:0]
    g1[6] = 0; g1[7] = 0;
    v4i z = {0, 0, 0, 0};
#if __clang_major__ >= 23
    v8i z8 = {0, 0, 0, 0, 0, 0, 0, 0};
    __builtin_amdgcn_tensor_load_to_lds(g0, g1, z, z, z8, 0);
#else
    __builtin_amdgcn_tensor_load_to_lds(g0, g1, z, z, 0);
#endif
}
#else
#define HAVE_TDM 0
#endif

// ---------------- fused projection GEMM (bf16 WMMA, TDM-staged) ----------------
// C[20000,2048] = Xb[20000,256] @ Bcat[256,2048]  (B stored transposed as BcatT)
// cols [0,1024)    -> featb (bf16)
// cols [1024,2048) -> acc = res + bias (f32)

__global__ __launch_bounds__(256)
void gemm_wmma_kernel(const __bf16* __restrict__ Xb, const __bf16* __restrict__ BcatT,
                      const float* __restrict__ bias,
                      __bf16* __restrict__ featb, float* __restrict__ acc) {
    // single shared object -> known LDS offsets: A slab at 0, B slab at BM*LDSK*2
    __shared__ __bf16 sm[(BM + BN) * LDSK];            // ~149 KB of the 320 KB WGP LDS

    const int tid  = threadIdx.x;
    const int wave = tid >> 5, lane = tid & 31;
    const int half = lane >> 4, mr = lane & 15;
    const int wm = wave >> 2, wn = wave & 3;           // 2 (M) x 4 (N) waves
    const int m0 = blockIdx.y * BM;
    const int n0 = blockIdx.x * BN;

#if HAVE_TDM
    if (tid < 32) {                                    // wave 0 drives both DMAs
        tdm_load_2d(0u, Xb + (size_t)m0 * GK, GK, BM, GK, GN, GK);
        tdm_load_2d((unsigned)(BM * LDSK * 2), BcatT + (size_t)n0 * GK, GK, BN, GK, GCOLS, GK);
        __builtin_amdgcn_s_wait_tensorcnt(0);
    }
    __syncthreads();
    // The TDM wrote LDS behind an opaque descriptor; tell the compiler sm is live
    // (escape the pointer + memory clobber) so the ds_load_b128 fragment reads survive.
    asm volatile("" :: "s"(&sm[0]) : "memory");
#else
    for (int ch = tid; ch < (BM * GK) / 8; ch += 256) {
        int row = ch >> 5, k8 = (ch & 31) * 8;
        *(uint4*)&sm[row * LDSK + k8] = *(const uint4*)(Xb + (size_t)(m0 + row) * GK + k8);
    }
    for (int ch = tid; ch < (BN * GK) / 8; ch += 256) {
        int row = ch >> 5, k8 = (ch & 31) * 8;
        *(uint4*)&sm[BM * LDSK + row * LDSK + k8] =
            *(const uint4*)(BcatT + (size_t)(n0 + row) * GK + k8);
    }
    __syncthreads();
#endif

    v8f c[4] = {};
    const int arow = wm * 16 + mr;
    const int khi  = half * 8;                         // lane's K sub-offset per ISA layout

    #pragma unroll
    for (int k0 = 0; k0 < GK; k0 += 32) {
        v8bf alo = *(const v8bf*)&sm[arow * LDSK + k0 + khi];
        v8bf ahi = *(const v8bf*)&sm[arow * LDSK + k0 + 16 + khi];
        v16bf a = __builtin_shufflevector(alo, ahi, 0,1,2,3,4,5,6,7,8,9,10,11,12,13,14,15);
        #pragma unroll
        for (int t = 0; t < 4; ++t) {
            int brow = wn * 64 + t * 16 + mr;
            v8bf blo = *(const v8bf*)&sm[BM * LDSK + brow * LDSK + k0 + khi];
            v8bf bhi = *(const v8bf*)&sm[BM * LDSK + brow * LDSK + k0 + 16 + khi];
            v16bf b = __builtin_shufflevector(blo, bhi, 0,1,2,3,4,5,6,7,8,9,10,11,12,13,14,15);
            c[t] = __builtin_amdgcn_wmma_f32_16x16x32_bf16(false, a, false, b,
                                                           (short)0, c[t], false, false);
        }
    }

    // C layout: VGPR r, lanes 0-15: M=r, lanes 16-31: M=8+r; N=mr
    #pragma unroll
    for (int t = 0; t < 4; ++t) {
        int col = n0 + wn * 64 + t * 16 + mr;
        #pragma unroll
        for (int r = 0; r < 8; ++r) {
            int row = m0 + wm * 16 + half * 8 + r;
            float v = c[t][r];
            if (col < GHD) {
                featb[(size_t)row * GHD + col] = (__bf16)v;
            } else {
                int cc = col - GHD;
                acc[(size_t)row * GHD + cc] = v + bias[cc];
            }
        }
    }
}

// ---------------- attention scores ----------------

__global__ void attn_scores_kernel(const __bf16* __restrict__ featb,
                                   const float* __restrict__ attn_l, const float* __restrict__ attn_r,
                                   float* __restrict__ el, float* __restrict__ er) {
    int t = blockIdx.x * blockDim.x + threadIdx.x;   // t = n*8 + h
    if (t >= GN * GH) return;
    int h = t & 7;
    const __bf16* fp = featb + (size_t)t * GD;
    const float* al = attn_l + h * GD;
    const float* ar = attn_r + h * GD;
    float sl = 0.f, sr = 0.f;
    #pragma unroll 4
    for (int d = 0; d < GD; ++d) { float f = (float)fp[d]; sl += f * al[d]; sr += f * ar[d]; }
    el[t] = sl; er[t] = sr;
}

// ---------------- edge softmax passes ----------------

__device__ __forceinline__ float leaky02(float v) { return v > 0.f ? v : 0.2f * v; }

__device__ __forceinline__ void atomicMaxF32(float* addr, float val) {
    if (val >= 0.f) atomicMax((int*)addr, __float_as_int(val));
    else            atomicMin((unsigned int*)addr, __float_as_uint(val));
}

__global__ void edge_max_kernel(const long long* __restrict__ src, const long long* __restrict__ dst,
                                const float* __restrict__ el, const float* __restrict__ er,
                                float* __restrict__ m, int E) {
    int t = blockIdx.x * blockDim.x + threadIdx.x;
    if (t >= E * GH) return;
    int e = t >> 3, h = t & 7;
    int s = (int)src[e], d = (int)dst[e];
    float v = leaky02(el[s * GH + h] + er[d * GH + h]);
    atomicMaxF32(&m[d * GH + h], v);
}

__global__ void edge_sum_kernel(const long long* __restrict__ src, const long long* __restrict__ dst,
                                const float* __restrict__ el, const float* __restrict__ er,
                                const float* __restrict__ m, float* __restrict__ ssum, int E) {
    int t = blockIdx.x * blockDim.x + threadIdx.x;
    if (t >= E * GH) return;
    int e = t >> 3, h = t & 7;
    int s = (int)src[e], d = (int)dst[e];
    float v = leaky02(el[s * GH + h] + er[d * GH + h]);
    atomicAdd(&ssum[d * GH + h], __expf(v - m[d * GH + h]));
}

// one block (256 threads) per edge: 1024 elems, 4/thread; alphas staged in LDS
__global__ __launch_bounds__(256)
void edge_scatter_kernel(const long long* __restrict__ src, const long long* __restrict__ dst,
                         const float* __restrict__ el, const float* __restrict__ er,
                         const float* __restrict__ m, const float* __restrict__ ssum,
                         const __bf16* __restrict__ featb, float* __restrict__ acc) {
    __shared__ float alph[GH];
    int e = blockIdx.x;
    int s = (int)src[e], d = (int)dst[e];
    if (threadIdx.x < GH) {
        int h = threadIdx.x;
        float v = leaky02(el[s * GH + h] + er[d * GH + h]);
        alph[h] = __expf(v - m[d * GH + h]) / ssum[d * GH + h];
    }
    __syncthreads();
    int base = threadIdx.x * 4;
    const __bf16* fp = featb + (size_t)s * GHD + base;
    float*        ap = acc   + (size_t)d * GHD + base;
    #pragma unroll
    for (int u = 0; u < 4; ++u) {
        int h = (base + u) >> 7;
        atomicAdd(&ap[u], alph[h] * (float)fp[u]);
    }
}

// ---------------- head-sum + ReLU + BN stats ----------------

__global__ __launch_bounds__(128)
void headsum_kernel(const float* __restrict__ acc, float* __restrict__ hbuf, float* __restrict__ stats) {
    int n = blockIdx.x, d = threadIdx.x;   // blockDim = 128
    const float* p = acc + (size_t)n * GHD + d;
    float sum = 0.f;
    #pragma unroll
    for (int h = 0; h < GH; ++h) sum += p[h * GD];
    sum = sum > 0.f ? sum : 0.f;
    hbuf[(size_t)n * GD + d] = sum;
    atomicAdd(&stats[d], sum);             // mean accumulator
    atomicAdd(&stats[GD + d], sum * sum);  // E[x^2] accumulator
}

__global__ void bn_finalize_kernel(const float* __restrict__ stats,
                                   const float* __restrict__ gamma, const float* __restrict__ beta,
                                   float* __restrict__ bnscale, float* __restrict__ bnshift) {
    int d = threadIdx.x;
    float mean = stats[d] * (1.0f / (float)GN);
    float var  = stats[GD + d] * (1.0f / (float)GN) - mean * mean;
    float sc = gamma[d] * rsqrtf(var + 1e-5f);
    bnscale[d] = sc;
    bnshift[d] = beta[d] - mean * sc;
}

// ---------------- BN apply + mu / log_sigma heads ----------------

__global__ __launch_bounds__(128)
void mlp_kernel(const float* __restrict__ hbuf, const float* __restrict__ bnscale,
                const float* __restrict__ bnshift,
                const float* __restrict__ muW, const float* __restrict__ mub,
                const float* __restrict__ lsW, const float* __restrict__ lsb,
                float* __restrict__ out) {
    __shared__ float hs[GD];
    int n = blockIdx.x, t = threadIdx.x;   // blockDim = 128
    hs[t] = hbuf[(size_t)n * GD + t] * bnscale[t] + bnshift[t];
    __syncthreads();
    int j = t & 63;
    const float* Wp = (t < GZ) ? muW : lsW;
    float sum = (t < GZ) ? mub[j] : lsb[j];
    #pragma unroll 4
    for (int d = 0; d < GD; ++d) sum += hs[d] * Wp[d * GZ + j];
    if (t < GZ) {
        out[(size_t)n * GZ + j] = sum;
    } else {
        sum = fminf(fmaxf(sum, -10.f), 10.f);
        out[(size_t)GN * GZ + (size_t)n * GZ + j] = sum;
    }
}

// ---------------- host launcher ----------------

extern "C" void kernel_launch(void* const* d_in, const int* in_sizes, int n_in,
                              void* d_out, int out_size, void* d_ws, size_t ws_size,
                              hipStream_t stream) {
    const float* x      = (const float*)d_in[0];
    const float* W      = (const float*)d_in[1];
    const float* attn_l = (const float*)d_in[2];
    const float* attn_r = (const float*)d_in[3];
    const float* res_W  = (const float*)d_in[4];
    const float* bias   = (const float*)d_in[5];
    const float* gamma  = (const float*)d_in[6];
    const float* beta   = (const float*)d_in[7];
    const float* mu_W   = (const float*)d_in[8];
    const float* mu_b   = (const float*)d_in[9];
    const float* ls_W   = (const float*)d_in[10];
    const float* ls_b   = (const float*)d_in[11];
    const long long* src = (const long long*)d_in[12];
    const long long* dst = (const long long*)d_in[13];
    const int E = in_sizes[12];
    float* out = (float*)d_out;

    // workspace carve-up (256B aligned)
    char* base = (char*)d_ws;
    size_t off = 0;
    auto carve = [&](size_t bytes) { char* p = base + off; off = (off + bytes + 255) & ~(size_t)255; return p; };
    __bf16* Xb      = (__bf16*)carve((size_t)GN * GK * sizeof(__bf16));
    __bf16* BcatT   = (__bf16*)carve((size_t)GK * GCOLS * sizeof(__bf16));
    __bf16* featb   = (__bf16*)carve((size_t)GN * GHD * sizeof(__bf16));
    float*  acc     = (float*) carve((size_t)GN * GHD * sizeof(float));
    float*  el      = (float*) carve((size_t)GN * GH * sizeof(float));
    float*  er      = (float*) carve((size_t)GN * GH * sizeof(float));
    float*  mbuf    = (float*) carve((size_t)GN * GH * sizeof(float));
    float*  ssum    = (float*) carve((size_t)GN * GH * sizeof(float));
    float*  hbuf    = (float*) carve((size_t)GN * GD * sizeof(float));
    float*  stats   = (float*) carve((size_t)2 * GD * sizeof(float));
    float*  bnscale = (float*) carve((size_t)GD * sizeof(float));
    float*  bnshift = (float*) carve((size_t)GD * sizeof(float));
    (void)ws_size; (void)n_in; (void)out_size;

    // 1. precision conversion + init
    convert_x_kernel<<<(GN * GK + 255) / 256, 256, 0, stream>>>(x, Xb, GN * GK);
    convert_B_kernel<<<(GK * GCOLS + 255) / 256, 256, 0, stream>>>(W, res_W, BcatT);
    init_kernel<<<(GN * GH + 255) / 256, 256, 0, stream>>>(mbuf, ssum, stats);

    // 2. fused projection GEMM on the matrix units (TDM-staged tiles)
    dim3 ggrid(GCOLS / BN, GN / BM);
    gemm_wmma_kernel<<<ggrid, 256, 0, stream>>>(Xb, BcatT, bias, featb, acc);

    // 3. attention scores
    attn_scores_kernel<<<(GN * GH + 255) / 256, 256, 0, stream>>>(featb, attn_l, attn_r, el, er);

    // 4-6. edge softmax + weighted scatter
    int egrid = (E * GH + 255) / 256;
    edge_max_kernel<<<egrid, 256, 0, stream>>>(src, dst, el, er, mbuf, E);
    edge_sum_kernel<<<egrid, 256, 0, stream>>>(src, dst, el, er, mbuf, ssum, E);
    edge_scatter_kernel<<<E, 256, 0, stream>>>(src, dst, el, er, mbuf, ssum, featb, acc);

    // 7-9. head-sum + BN + latent heads
    headsum_kernel<<<GN, 128, 0, stream>>>(acc, hbuf, stats);
    bn_finalize_kernel<<<1, 128, 0, stream>>>(stats, gamma, beta, bnscale, bnshift);
    mlp_kernel<<<GN, 128, 0, stream>>>(hbuf, bnscale, bnshift, mu_W, mu_b, ls_W, ls_b, out);
}
